// KDALayer_79688823210385
// MI455X (gfx1250) — matrix operand order repaired and checked by reading, not compiled, not run on hardware
//
#include <hip/hip_runtime.h>
#include <hip/hip_bf16.h>
#include <cstdint>
#include <cstddef>

// ---------------------------------------------------------------------------
// KDA layer for MI455X (gfx1250).
// All GEMM-shaped stages use v_wmma_f32_16x16x32_bf16 (f32 accumulate);
// tile staging uses CDNA5 async global->LDS DMA (ASYNCcnt) with double
// buffering so DMA overlaps WMMA.
// B=2, T=2048, D=2048, NH=16, HD=128.
// ---------------------------------------------------------------------------

typedef __attribute__((ext_vector_type(16))) __bf16 bf16x16;
typedef __attribute__((ext_vector_type(8)))  __bf16 bf16x8;   // 16 bytes
typedef __attribute__((ext_vector_type(8)))  float  floatx8;

static __device__ __forceinline__ floatx8 wmma_bf16(bf16x16 a, bf16x16 b, floatx8 c) {
  return __builtin_amdgcn_wmma_f32_16x16x32_bf16(false, a, false, b, (short)0, c,
                                                 false, false);
}

// CDNA5 async global->LDS copy, 16 bytes per lane. lds_off = workgroup-relative
// LDS byte offset (low 32 bits of a generic pointer to __shared__).
static __device__ __forceinline__ void async_b128(uint32_t lds_off, const void* gaddr) {
  asm volatile("global_load_async_to_lds_b128 %0, %1, off"
               :: "v"(lds_off), "v"(gaddr) : "memory");
}
static __device__ __forceinline__ void wait_async0() {
  asm volatile("s_wait_asynccnt 0x0" ::: "memory");
}
static __device__ __forceinline__ uint32_t lds_off32(const void* p) {
  return (uint32_t)(uintptr_t)p;
}

// A-fragment (16x32 bf16, row-major source; lane r = row, K split hi*8 / 16+hi*8)
static __device__ __forceinline__ bf16x16 frag_ld_a(const __bf16* buf, int row0,
                                                    int stride, int k0) {
  const int lane = threadIdx.x & 31;
  const int r = lane & 15, hi = lane >> 4;
  const __bf16* p = buf + (size_t)(row0 + r) * stride + k0 + hi * 8;
  bf16x8 lo = *(const bf16x8*)(p);
  bf16x8 hi8 = *(const bf16x8*)(p + 16);
  bf16x16 out;
#pragma unroll
  for (int i = 0; i < 8; ++i) { out[i] = lo[i]; out[8 + i] = hi8[i]; }
  return out;
}

// B-fragment (32x16 bf16). B = S^T where S is row-major [n][k]; lane n = column,
// K split hi*8 / 16+hi*8.
static __device__ __forceinline__ bf16x16 frag_ld_b(const __bf16* buf, int row0,
                                                    int stride, int k0) {
  const int lane = threadIdx.x & 31;
  const int n = lane & 15, hi = lane >> 4;
  const __bf16* p = buf + (size_t)(row0 + n) * stride + k0 + hi * 8;
  bf16x8 lo = *(const bf16x8*)(p);
  bf16x8 hi8 = *(const bf16x8*)(p + 16);
  bf16x16 out;
#pragma unroll
  for (int i = 0; i < 8; ++i) { out[i] = lo[i]; out[8 + i] = hi8[i]; }
  return out;
}

// ---------------------------------------------------------------------------
// f32 -> bf16 conversion
// ---------------------------------------------------------------------------
__global__ void cvt_bf16_kernel(const float* __restrict__ src,
                                __bf16* __restrict__ dst, size_t n) {
  size_t i = (size_t)blockIdx.x * blockDim.x + threadIdx.x;
  size_t stride = (size_t)gridDim.x * blockDim.x;
  for (; i < n; i += stride) dst[i] = (__bf16)src[i];
}

// ---------------------------------------------------------------------------
// prep: alpha = sigmoid(alpha_log); log_a[h] = log(clip(mean)); cos/sin tables
// ---------------------------------------------------------------------------
__global__ void prep_kernel(const float* __restrict__ alpha_log,
                            float* __restrict__ alpha, float* __restrict__ log_a,
                            float* __restrict__ ctab, float* __restrict__ stab) {
  int tid = threadIdx.x;
  for (int i = tid; i < 2048; i += 256)
    alpha[i] = 1.0f / (1.0f + __expf(-alpha_log[i]));
  __syncthreads();
  if (tid < 16) {
    float s = 0.0f;
    for (int d = 0; d < 128; ++d) s += alpha[tid * 128 + d];
    log_a[tid] = logf(fmaxf(s / 128.0f, 1e-6f));
  }
  for (int i = tid; i < 1024; i += 256) {
    int hh = i >> 6, d = i & 63;
    float f = (float)hh * powf(10000.0f, -((float)d) / 64.0f);
    ctab[i] = cosf(f);
    stab[i] = sinf(f);
  }
}

// ---------------------------------------------------------------------------
// beta[b,h,t] = sigmoid(x[row] . Wb[h] + bb[h]),  stored head-major (B*NH, T)
// ---------------------------------------------------------------------------
__global__ __launch_bounds__(256) void beta_kernel(
    const float* __restrict__ x, const float* __restrict__ Wb,
    const float* __restrict__ bbias, float* __restrict__ betaT, int T_, int D_) {
  __shared__ float red[256];
  const int row = blockIdx.x;            // b*T + t
  const int b = row / T_, t = row % T_;
  const int tid = threadIdx.x;
  const int hh = tid >> 4;               // 16 heads
  const int p = tid & 15;
  const float* xr = x + (size_t)row * D_;
  const float* wr = Wb + (size_t)hh * D_;
  float sum = 0.0f;
  for (int k = p; k < D_; k += 16) sum += xr[k] * wr[k];
  red[tid] = sum;
  __syncthreads();
  if (p == 0) {
    float s = 0.0f;
    for (int i = 0; i < 16; ++i) s += red[hh * 16 + i];
    s += bbias[hh];
    betaT[((size_t)(b * 16 + hh)) * T_ + t] = 1.0f / (1.0f + __expf(-s));
  }
}

// ---------------------------------------------------------------------------
// GEMM: C[m,n] = sum_k A[m,k] * W[n,k]   (A: MxK bf16 rm, W: NxK bf16 rm, C f32)
// block tile 128x128, K-step 32; 8 waves = 4(m) x 2(n); wave tile 32x64.
// Async double-buffered staging: DMA of tile k+1 overlaps WMMA on tile k.
// ---------------------------------------------------------------------------
#define GEMM_LDSS 40
__global__ __launch_bounds__(256) void gemm_bt_kernel(
    const __bf16* __restrict__ A, const __bf16* __restrict__ W,
    float* __restrict__ C, int M, int N, int K) {
  __shared__ __align__(16) __bf16 As[2][128 * GEMM_LDSS];
  __shared__ __align__(16) __bf16 Bs[2][128 * GEMM_LDSS];
  const int m0 = blockIdx.x * 128;
  const int n0 = blockIdx.y * 128;
  const int tid = threadIdx.x;
  const int wave = tid >> 5;
  const int wm = wave & 3;     // m sub-block of 32 rows
  const int wn = wave >> 2;    // n sub-block of 64 cols
  const int lrow = tid >> 1;             // 0..127
  const int lseg = (tid & 1) * 16;       // halves

  auto stage = [&](int k0, int bufi) {
    const __bf16* ga = A + (size_t)(m0 + lrow) * K + k0 + lseg;
    const __bf16* gw = W + (size_t)(n0 + lrow) * K + k0 + lseg;
    uint32_t la = lds_off32(&As[bufi][lrow * GEMM_LDSS + lseg]);
    uint32_t lb = lds_off32(&Bs[bufi][lrow * GEMM_LDSS + lseg]);
    async_b128(la,      ga);
    async_b128(la + 16, ga + 8);
    async_b128(lb,      gw);
    async_b128(lb + 16, gw + 8);
  };

  floatx8 acc[2][4] = {};
  const int nk = K >> 5;
  stage(0, 0);
  for (int kt = 0; kt < nk; ++kt) {
    wait_async0();                      // own tile-kt DMAs complete
    __syncthreads();                    // everyone's tile-kt data visible
    if (kt + 1 < nk) stage((kt + 1) << 5, (kt + 1) & 1);
    const __bf16* as = As[kt & 1];
    const __bf16* bs = Bs[kt & 1];
    bf16x16 af[2], bfrag[4];
#pragma unroll
    for (int i = 0; i < 2; ++i) af[i] = frag_ld_a(as, wm * 32 + i * 16, GEMM_LDSS, 0);
#pragma unroll
    for (int j = 0; j < 4; ++j) bfrag[j] = frag_ld_b(bs, wn * 64 + j * 16, GEMM_LDSS, 0);
#pragma unroll
    for (int i = 0; i < 2; ++i)
#pragma unroll
      for (int j = 0; j < 4; ++j)
        acc[i][j] = wmma_bf16(af[i], bfrag[j], acc[i][j]);
  }
  const int lane = tid & 31;
  const int nlo = lane & 15, mhi = lane >> 4;
#pragma unroll
  for (int i = 0; i < 2; ++i)
    for (int j = 0; j < 4; ++j)
#pragma unroll
      for (int r = 0; r < 8; ++r) {
        int m = m0 + wm * 32 + i * 16 + r + 8 * mhi;
        int n = n0 + wn * 64 + j * 16 + nlo;
        C[(size_t)m * N + n] = acc[i][j][r];
      }
}

// ---------------------------------------------------------------------------
// Per-head rotation of q,k; pack to bf16 head-major [B,NH,T,HD]
// qkv layout: [B*T, 6144] = [q(2048) | k(2048) | v(2048)]
// ---------------------------------------------------------------------------
__global__ __launch_bounds__(256) void rot_pack_kernel(
    const float* __restrict__ qkv, const float* __restrict__ ctab,
    const float* __restrict__ stab, __bf16* __restrict__ Qh,
    __bf16* __restrict__ Kh, int T_) {
  const int row = blockIdx.x;            // b*T + t
  const int b = row / T_, t = row % T_;
  const int tid = threadIdx.x;
  const float* src = qkv + (size_t)row * 6144;
#pragma unroll
  for (int which = 0; which < 2; ++which) {
    const float* s_ = src + which * 2048;
    __bf16* dst = (which == 0) ? Qh : Kh;
    for (int idx = tid; idx < 1024; idx += 256) {
      int hh = idx >> 6, d0 = idx & 63;
      float u1 = s_[hh * 128 + d0];
      float u2 = s_[hh * 128 + 64 + d0];
      float c = ctab[hh * 64 + d0], sn = stab[hh * 64 + d0];
      size_t base = ((size_t)(b * 16 + hh) * T_ + t) * 128;
      dst[base + d0]      = (__bf16)(u1 * c - u2 * sn);
      dst[base + 64 + d0] = (__bf16)(u2 * c + u1 * sn);
    }
  }
}

// ---------------------------------------------------------------------------
// V transpose + pack: VT[b,h,e,t] (bf16), via LDS tile  (64 t x 128 e per block)
// ---------------------------------------------------------------------------
__global__ __launch_bounds__(256) void pack_vT_kernel(
    const float* __restrict__ qkv, __bf16* __restrict__ VT, int T_) {
  __shared__ __bf16 tile[128 * 66];
  const int bh = blockIdx.x;
  const int b = bh >> 4, h = bh & 15;
  const int t0 = blockIdx.y * 64;
  const int tid = threadIdx.x;
  for (int l = tid; l < 8192; l += 256) {
    int tt = l >> 7, e = l & 127;
    float v = qkv[((size_t)(b * T_ + t0 + tt)) * 6144 + 4096 + h * 128 + e];
    tile[e * 66 + tt] = (__bf16)v;
  }
  __syncthreads();
  for (int l = tid; l < 8192; l += 256) {
    int e = l >> 6, tt = l & 63;
    VT[((size_t)bh * 128 + e) * T_ + t0 + tt] = tile[e * 66 + tt];
  }
}

// ---------------------------------------------------------------------------
// Attention: per (b,h), 128-row t-tile; 8 waves x 16 rows. Two WMMA stages:
//   S = Q K^T  (K=128),  then  O += (S * decay * beta) V  (K over s)
// Async double-buffered K/VT staging; causal chunks skipped wave-uniformly
// (EXEC stays all-ones around WMMA, barrier schedule uniform).
// ---------------------------------------------------------------------------
__global__ __launch_bounds__(256) void attn_kernel(
    const __bf16* __restrict__ Q, const __bf16* __restrict__ Kh,
    const __bf16* __restrict__ VT, const float* __restrict__ betaT,
    const float* __restrict__ log_a, __bf16* __restrict__ Ob, int T_) {
  const int KS_S = 136, VT_S = 40, P_S = 40;
  __shared__ __align__(16) __bf16 Ks[2][32 * 136];
  __shared__ __align__(16) __bf16 VTs[2][128 * 40];
  __shared__ __align__(16) __bf16 Ps[8 * 16 * 40];
  const int bh = blockIdx.y;             // b*NH + h
  const int h = bh & 15, b = bh >> 4;
  const int t0 = blockIdx.x * 128;
  const int tid = threadIdx.x;
  const int wave = tid >> 5, lane = tid & 31;
  const int trow0 = t0 + wave * 16;
  const int nlo = lane & 15, mhi = lane >> 4;
  const float la = log_a[h];
  const __bf16* Qb  = Q  + (size_t)bh * T_ * 128;
  const __bf16* Kb  = Kh + (size_t)bh * T_ * 128;
  const __bf16* VTb = VT + (size_t)bh * 128 * T_;
  const float* betp = betaT + (size_t)bh * T_;

  auto stage = [&](int it, int bufi) {
    const int s0 = it * 32;
    int row = tid >> 3, seg = (tid & 7) * 16;
    const __bf16* g = Kb + (size_t)(s0 + row) * 128 + seg;
    uint32_t lk = lds_off32(&Ks[bufi][row * KS_S + seg]);
    async_b128(lk,      g);
    async_b128(lk + 16, g + 8);
    int vrow = tid >> 1, vseg = (tid & 1) * 16;
    const __bf16* gv = VTb + (size_t)vrow * T_ + s0 + vseg;
    uint32_t lv = lds_off32(&VTs[bufi][vrow * VT_S + vseg]);
    async_b128(lv,      gv);
    async_b128(lv + 16, gv + 8);
  };

  bf16x16 qf[4];
#pragma unroll
  for (int d = 0; d < 4; ++d) qf[d] = frag_ld_a(Qb, trow0, 128, d * 32);

  floatx8 o[8] = {};
  __bf16* Pw = &Ps[wave * 16 * P_S];

  const int nIter = (t0 + 128) >> 5;
  stage(0, 0);
  for (int it = 0; it < nIter; ++it) {
    const int s0 = it * 32;
    wait_async0();
    __syncthreads();
    if (it + 1 < nIter) stage(it + 1, (it + 1) & 1);
    const __bf16* KsC  = Ks[it & 1];
    const __bf16* VTsC = VTs[it & 1];
    if (s0 <= trow0 + 15) {              // wave-uniform causal skip
#pragma unroll
      for (int ss = 0; ss < 32; ss += 16) {
        floatx8 sacc = {};
#pragma unroll
        for (int d = 0; d < 4; ++d)
          sacc = wmma_bf16(qf[d], frag_ld_b(KsC, ss, KS_S, d * 32), sacc);
        int s = s0 + ss + nlo;
        float bet = betp[s];
#pragma unroll
        for (int r = 0; r < 8; ++r) {
          int t = trow0 + r + 8 * mhi;
          float w = (s <= t) ? sacc[r] * __expf((float)(t - s) * la) * bet : 0.0f;
          Pw[(r + 8 * mhi) * P_S + ss + nlo] = (__bf16)w;
        }
      }
      bf16x16 pf = frag_ld_a(Pw, 0, P_S, 0);
#pragma unroll
      for (int nf = 0; nf < 8; ++nf)
        o[nf] = wmma_bf16(pf, frag_ld_b(VTsC, nf * 16, VT_S, 0), o[nf]);
    }
  }
  // Ob[b, t, h*128 + e]  (bf16, input to the output-projection GEMM)
#pragma unroll
  for (int nf = 0; nf < 8; ++nf)
#pragma unroll
    for (int r = 0; r < 8; ++r) {
      int t = trow0 + r + 8 * mhi;
      int e = nf * 16 + nlo;
      Ob[((size_t)(b * T_ + t)) * 2048 + h * 128 + e] = (__bf16)o[nf][r];
    }
}

// ---------------------------------------------------------------------------
// State (closed form of the scan):
//   state[b,h,i,j] = sum_t alpha[h,i]^(T-1-t) * beta[b,t,h] * kh[t,i] * v[t,j]
// Build w^T[i, t-chunk] in LDS (VALU) while VT chunk streams in via async DMA,
// then WMMA.  One block per (b,h).
// ---------------------------------------------------------------------------
__global__ __launch_bounds__(256) void state_kernel(
    const __bf16* __restrict__ Kh, const __bf16* __restrict__ VT,
    const float* __restrict__ betaT, const float* __restrict__ alpha,
    float* __restrict__ stateOut, int T_) {
  __shared__ __align__(16) __bf16 Wt[128 * 40];
  __shared__ __align__(16) __bf16 VTs[128 * 40];
  __shared__ float lalpha[128];
  const int bh = blockIdx.x;
  const int h = bh & 15;
  const int tid = threadIdx.x, wave = tid >> 5, lane = tid & 31;
  if (tid < 128) lalpha[tid] = logf(alpha[h * 128 + tid]);
  __syncthreads();
  const __bf16* Kb  = Kh + (size_t)bh * T_ * 128;
  const __bf16* VTb = VT + (size_t)bh * 128 * T_;
  const float* betp = betaT + (size_t)bh * T_;
  floatx8 acc[8] = {};
  for (int s0 = 0; s0 < T_; s0 += 32) {
    {   // async DMA the VT chunk first; overlaps with the VALU tile build below
      int vrow = tid >> 1, vseg = (tid & 1) * 16;
      const __bf16* gv = VTb + (size_t)vrow * T_ + s0 + vseg;
      uint32_t lv = lds_off32(&VTs[vrow * 40 + vseg]);
      async_b128(lv,      gv);
      async_b128(lv + 16, gv + 8);
    }
#pragma unroll
    for (int rep = 0; rep < 16; ++rep) {
      int linear = tid + rep * 256;      // 0..4095
      int i = linear & 127, tt = linear >> 7;
      int t = s0 + tt;
      float w = __expf((float)(T_ - 1 - t) * lalpha[i]) * betp[t] *
                (float)Kb[(size_t)t * 128 + i];
      Wt[i * 40 + tt] = (__bf16)w;
    }
    wait_async0();
    __syncthreads();
    bf16x16 wf = frag_ld_a(Wt, wave * 16, 40, 0);
#pragma unroll
    for (int nf = 0; nf < 8; ++nf)
      acc[nf] = wmma_bf16(wf, frag_ld_b(VTs, nf * 16, 40, 0), acc[nf]);
    __syncthreads();
  }
  const int nlo = lane & 15, mhi = lane >> 4;
#pragma unroll
  for (int nf = 0; nf < 8; ++nf)
    for (int r = 0; r < 8; ++r) {
      int i = wave * 16 + r + 8 * mhi;
      int j = nf * 16 + nlo;
      stateOut[(size_t)bh * 128 * 128 + (size_t)i * 128 + j] = acc[nf][r];
    }
}

// ---------------------------------------------------------------------------
// Host-side launcher
// ---------------------------------------------------------------------------
extern "C" void kernel_launch(void* const* d_in, const int* in_sizes, int n_in,
                              void* d_out, int out_size, void* d_ws, size_t ws_size,
                              hipStream_t stream) {
  (void)in_sizes; (void)n_in; (void)out_size; (void)ws_size;
  const int B_ = 2, T_ = 2048, D_ = 2048, NH_ = 16;
  const int M = B_ * T_;                 // 4096 rows
  const float* x         = (const float*)d_in[0];
  const float* Wq        = (const float*)d_in[1];
  const float* Wk        = (const float*)d_in[2];
  const float* Wv        = (const float*)d_in[3];
  const float* Wo        = (const float*)d_in[4];
  const float* Wb        = (const float*)d_in[5];
  const float* bbias     = (const float*)d_in[6];
  const float* alpha_log = (const float*)d_in[7];

  char* ws = (char*)d_ws;
  size_t off = 0;
  auto alloc = [&](size_t bytes) -> void* {
    void* p = ws + off;
    off = (off + bytes + 255) & ~(size_t)255;
    return p;
  };
  __bf16* xb    = (__bf16*)alloc((size_t)M * D_ * 2);           // 16.8 MB
  __bf16* wqkv  = (__bf16*)alloc((size_t)3 * D_ * D_ * 2);      // 25.2 MB
  __bf16* wob   = (__bf16*)alloc((size_t)D_ * D_ * 2);          //  8.4 MB
  float*  qkvf  = (float*) alloc((size_t)M * 3 * D_ * 4);       // 100.7 MB
  __bf16* qh    = (__bf16*)alloc((size_t)M * D_ * 2);
  __bf16* kh    = (__bf16*)alloc((size_t)M * D_ * 2);
  __bf16* vT    = (__bf16*)alloc((size_t)M * D_ * 2);
  __bf16* ob    = (__bf16*)alloc((size_t)M * D_ * 2);
  float*  betaT = (float*) alloc((size_t)B_ * NH_ * T_ * 4);
  float*  alpha = (float*) alloc(2048 * 4);
  float*  log_a = (float*) alloc(64);
  float*  ctab  = (float*) alloc(1024 * 4);
  float*  stab  = (float*) alloc(1024 * 4);

  // 1) precision conversion (bf16 operands for WMMA)
  cvt_bf16_kernel<<<2048, 256, 0, stream>>>(x,  xb,   (size_t)M * D_);
  cvt_bf16_kernel<<<1024, 256, 0, stream>>>(Wq, wqkv,                       (size_t)D_ * D_);
  cvt_bf16_kernel<<<1024, 256, 0, stream>>>(Wk, wqkv + (size_t)D_ * D_,     (size_t)D_ * D_);
  cvt_bf16_kernel<<<1024, 256, 0, stream>>>(Wv, wqkv + (size_t)2 * D_ * D_, (size_t)D_ * D_);
  cvt_bf16_kernel<<<1024, 256, 0, stream>>>(Wo, wob,  (size_t)D_ * D_);

  // 2) small tables + beta projection
  prep_kernel<<<1, 256, 0, stream>>>(alpha_log, alpha, log_a, ctab, stab);
  beta_kernel<<<M, 256, 0, stream>>>(x, Wb, bbias, betaT, T_, D_);

  // 3) fused QKV GEMM: (4096 x 2048) x (6144 x 2048)^T  -> qkvf
  gemm_bt_kernel<<<dim3(M / 128, (3 * D_) / 128), 256, 0, stream>>>(
      xb, wqkv, qkvf, M, 3 * D_, D_);

  // 4) rotate q,k + pack head-major bf16; transpose v
  rot_pack_kernel<<<M, 256, 0, stream>>>(qkvf, ctab, stab, qh, kh, T_);
  pack_vT_kernel<<<dim3(B_ * NH_, T_ / 64), 256, 0, stream>>>(qkvf, vT, T_);

  // 5) causal decayed attention -> ob (bf16, [B,T,NH*HD])
  attn_kernel<<<dim3(T_ / 128, B_ * NH_), 256, 0, stream>>>(
      qh, kh, vT, betaT, log_a, ob, T_);

  // 6) recurrent state in closed form -> second output
  state_kernel<<<B_ * NH_, 256, 0, stream>>>(
      kh, vT, betaT, alpha, (float*)d_out + (size_t)M * D_, T_);

  // 7) output projection -> first output (f32)
  gemm_bt_kernel<<<dim3(M / 128, D_ / 128), 256, 0, stream>>>(
      ob, wob, (float*)d_out, M, D_, D_);
}
